// ReferenceEncoder_16269336117552
// MI455X (gfx1250) — matrix operand order repaired and checked by reading, not compiled, not run on hardware
//
#include <hip/hip_runtime.h>
#include <hip/hip_bf16.h>

// ---------------------------------------------------------------------------
// ReferenceEncoder for MI455X (gfx1250): conv stack + BiLSTM x2 via WMMA bf16
// ---------------------------------------------------------------------------

#define NB 16
#define NT 2048
#define NL 512
#define NH 128
#define TPAD (NT + 2)   // padded time stride for conv activations

typedef __attribute__((ext_vector_type(16))) __bf16 v16bf;
typedef __attribute__((ext_vector_type(8)))  __bf16 v8bf;
typedef __attribute__((ext_vector_type(8)))  float  v8f;

union AF { v16bf v; v8bf h[2]; };

__device__ __forceinline__ float sigf(float x) { return 1.f / (1.f + __expf(-x)); }
__device__ __forceinline__ float tanhf_fast(float x) {
    x = fminf(15.f, fmaxf(-15.f, x));
    float e = __expf(2.f * x);
    return (e - 1.f) / (e + 1.f);
}

// ---------------------------------------------------------------------------
// Generic WMMA GEMM: Y = im2col(A) x Wb^T + bias
// A:  bf16 [B][Tstride][Cin]; logical row r=b*T+t reads taps t+dt-1 via the
//     zero-padded layout (caller passes A pre-shifted for KW==1).
// Wb: bf16 [Cout][KW*Cin] (row n = output channel, K contiguous).
// Branch-free, division-free inner loop: dt outer, ci inner.
// transOut: 0 -> Y[(b*T+t)*Cout+n], 1 -> Y[(t*16+b)*Cout+n] (time-major gates)
// ---------------------------------------------------------------------------
__global__ void wmma_gemm(const __bf16* __restrict__ A,
                          const __bf16* __restrict__ Wb,
                          const float*  __restrict__ bias,
                          float* __restrict__ Y,
                          int T, int Tstride, int Cin, int Cout, int KW,
                          int transOut)
{
    const int lane   = threadIdx.x & 31;
    const int wave   = threadIdx.x >> 5;
    const int strips = Cout >> 6;
    const long gw    = (long)blockIdx.x * 8 + wave;
    const int m_tile = (int)(gw / strips);
    const int n_strip= (int)(gw % strips);
    const int Ktot   = KW * Cin;

    const int rbase = m_tile * 16;        // T % 16 == 0 -> whole tile same b
    const int b     = rbase / T;
    const int tbase = rbase % T;
    const int t0    = tbase + (lane & 15);
    const int kh    = (lane < 16) ? 0 : 8;    // A-fragment K sub-offset
    const int kb    = (lane < 16) ? 0 : 8;    // B-fragment K sub-offset
    const int col0  = n_strip * 64 + (lane & 15);

    v8f acc[4];
#pragma unroll
    for (int nt = 0; nt < 4; nt++)
#pragma unroll
        for (int v = 0; v < 8; v++) acc[nt][v] = 0.f;

    for (int dt = 0; dt < KW; dt++) {
        const __bf16* arow = A + ((long)b * Tstride + t0 + dt) * Cin + kh;
        const __bf16* wrow = Wb + dt * Cin + kb;
        for (int ci = 0; ci < Cin; ci += 32) {
            AF a;
            a.h[0] = *(const v8bf*)(arow + ci);
            a.h[1] = *(const v8bf*)(arow + ci + 16);
#pragma unroll
            for (int nt = 0; nt < 4; nt++) {
                int n = col0 + nt * 16;
                const __bf16* wp = wrow + (long)n * Ktot + ci;
                AF bf;
                bf.h[0] = *(const v8bf*)(wp);
                bf.h[1] = *(const v8bf*)(wp + 16);
                acc[nt] = __builtin_amdgcn_wmma_f32_16x16x32_bf16(
                    false, a.v, false, bf.v, (short)0, acc[nt], false, false);
            }
        }
    }

    // ---- epilogue: bias add, f32 store ----
#pragma unroll
    for (int nt = 0; nt < 4; nt++) {
        int n = col0 + nt * 16;
        float bs = bias[n];
#pragma unroll
        for (int v = 0; v < 8; v++) {
            int m = ((lane < 16) ? 0 : 8) + v;
            long orow = transOut ? ((long)(tbase + m) * NB + b)
                                 : ((long)rbase + m);
            Y[orow * Cout + n] = acc[nt][v] + bs;
        }
    }
}

// ---------------------------------------------------------------------------
// Layer 0: fused input prep (log(pitch+1), energy) + conv1d(2->1024, k=3)
// ---------------------------------------------------------------------------
__global__ void conv0_kernel(const float* __restrict__ pitch,
                             const float* __restrict__ energy,
                             const float* __restrict__ w,   // [1024,2,3]
                             const float* __restrict__ bias,
                             float* __restrict__ Y)         // [B*T, 1024]
{
    int r = blockIdx.x;            // 0 .. B*T-1
    int b = r / NT, t = r % NT;
    float xin[3][2];
#pragma unroll
    for (int dt = 0; dt < 3; dt++) {
        int tt = t + dt - 1;
        if (tt >= 0 && tt < NT) {
            xin[dt][0] = logf(pitch[b * NT + tt] + 1.f);
            xin[dt][1] = energy[b * NT + tt];
        } else { xin[dt][0] = 0.f; xin[dt][1] = 0.f; }
    }
    for (int co = threadIdx.x; co < 1024; co += blockDim.x) {
        const float* wr = w + co * 6;       // [ci][dt]
        float acc = bias[co];
#pragma unroll
        for (int ci = 0; ci < 2; ci++)
#pragma unroll
            for (int dt = 0; dt < 3; dt++)
                acc += xin[dt][ci] * wr[ci * 3 + dt];
        Y[(long)r * 1024 + co] = acc;
    }
}

// ---------------------------------------------------------------------------
// BatchNorm training-mode stats, coalesced: block = 64 channels x 4 row-groups
// ---------------------------------------------------------------------------
__global__ void bn_stats(const float* __restrict__ Y, int rows, int C,
                         const float* __restrict__ gamma,
                         const float* __restrict__ beta,
                         float* __restrict__ scale, float* __restrict__ shift)
{
    int tx = threadIdx.x & 63, ty = threadIdx.x >> 6;
    int c = blockIdx.x * 64 + tx;
    float s = 0.f, s2 = 0.f;
    for (int r = ty; r < rows; r += 4) {
        float v = Y[(long)r * C + c];
        s += v; s2 += v * v;
    }
    __shared__ float sh[4][64], sh2[4][64];
    sh[ty][tx] = s; sh2[ty][tx] = s2;
    __syncthreads();
    if (ty == 0) {
        s  = sh[0][tx] + sh[1][tx] + sh[2][tx] + sh[3][tx];
        s2 = sh2[0][tx] + sh2[1][tx] + sh2[2][tx] + sh2[3][tx];
        float mean = s / (float)rows;
        float var  = s2 / (float)rows - mean * mean;
        float sc   = gamma[c] * rsqrtf(var + 1e-5f);
        scale[c] = sc;
        shift[c] = beta[c] - mean * sc;
    }
}

// BN+ReLU, f32 -> bf16 into the zero-padded [B][T+2][C] activation layout
__global__ void bn_apply_relu(const float* __restrict__ Y,
                              const float* __restrict__ scale,
                              const float* __restrict__ shift,
                              __bf16* __restrict__ out, long n, int C)
{
    long i = (long)blockIdx.x * blockDim.x + threadIdx.x;
    if (i < n) {
        int c = (int)(i % C);
        long r = i / C;
        int b = (int)(r / NT), t = (int)(r % NT);
        float v = Y[i] * scale[c] + shift[c];
        out[((long)b * TPAD + 1 + t) * C + c] = (__bf16)(v > 0.f ? v : 0.f);
    }
}

// zero the 2 pad rows per batch of a [B][T+2][C] activation buffer
__global__ void zero_pads(__bf16* __restrict__ act, int C)
{
    long i = (long)blockIdx.x * blockDim.x + threadIdx.x;
    long total = (long)32 * C;
    if (i < total) {
        int rowid = (int)(i / C), c = (int)(i % C);
        int b = rowid >> 1;
        long r = (long)b * TPAD + ((rowid & 1) ? (TPAD - 1) : 0);
        act[r * C + c] = (__bf16)0.f;
    }
}

// ---------------------------------------------------------------------------
// Conv weight reorder [Cout][Cin][3] f32 -> [Cout][dt*Cin+ci] bf16
// ---------------------------------------------------------------------------
__global__ void convw_reorder(const float* __restrict__ w, __bf16* __restrict__ Wb,
                              int Cout, int Cin)
{
    long total = (long)Cout * Cin * 3;
    long i = (long)blockIdx.x * blockDim.x + threadIdx.x;
    if (i < total) {
        int co  = (int)(i / ((long)Cin * 3));
        int rem = (int)(i % ((long)Cin * 3));
        int ci = rem / 3, dt = rem % 3;
        Wb[(long)co * (3 * Cin) + dt * Cin + ci] = (__bf16)w[i];
    }
}

// f32 [N][Ksrc] -> bf16 [N][Kdst], zero-padded K (also plain cvt)
__global__ void cvt_pad(const float* __restrict__ src, __bf16* __restrict__ dst,
                        int N, int Ksrc, int Kdst)
{
    long total = (long)N * Kdst;
    long i = (long)blockIdx.x * blockDim.x + threadIdx.x;
    if (i < total) {
        int n = (int)(i / Kdst), k = (int)(i % Kdst);
        dst[i] = (__bf16)((k < Ksrc) ? src[(long)n * Ksrc + k] : 0.f);
    }
}

// ---------------------------------------------------------------------------
// LSTM recurrence: one persistent workgroup per direction (blockIdx.x 0/1).
// Latency-bound: only 2 blocks on the whole GPU, so occupancy is irrelevant.
// W_hh B-fragments (16 x 8 VGPRs = 128 VGPRs) are preloaded into registers
// once; per step the critical path is gate loads (prefetched one step ahead
// via global_prefetch_b8) -> 16 WMMAs on LDS-resident h -> gate math.
// ---------------------------------------------------------------------------
__global__ void lstm_rec(const float* __restrict__ gatesF,
                         const float* __restrict__ gatesR,
                         const __bf16* __restrict__ WhhF,  // [512,128] bf16
                         const __bf16* __restrict__ WhhR,
                         float* __restrict__ Hout,         // [B*T, 256]
                         int T)
{
    const bool rev = (blockIdx.x == 1);
    const float*  gates = rev ? gatesR : gatesF;
    const __bf16* Whh   = rev ? WhhR   : WhhF;
    const int hoff = rev ? NH : 0;

    __shared__ float  gbuf[16][512];
    __shared__ float  cst[16][NH];
    __shared__ __bf16 hbf[16][NH];

    const int tid  = threadIdx.x;
    const int lane = tid & 31;
    const int wave = tid >> 5;

    for (int i = tid; i < 16 * NH; i += 256) {
        cst[i >> 7][i & 127] = 0.f;
        hbf[i >> 7][i & 127] = (__bf16)0.f;
    }
    __syncthreads();

    const int kh   = (lane < 16) ? 0 : 8;
    const int kb   = (lane < 16) ? 0 : 8;
    const int col0 = wave * 64 + (lane & 15);
    const int mrow = lane & 15;

    // ---- preload all W_hh B-fragments into registers (off critical path) ----
    AF bw[4][4];                          // [kk/32][nt]
#pragma unroll
    for (int k4 = 0; k4 < 4; k4++) {
#pragma unroll
        for (int nt = 0; nt < 4; nt++) {
            const __bf16* wp = Whh + (long)(col0 + nt * 16) * NH + k4 * 32 + kb;
            bw[k4][nt].h[0] = *(const v8bf*)(wp);
            bw[k4][nt].h[1] = *(const v8bf*)(wp + 16);
        }
    }

    for (int s = 0; s < T; s++) {
        int t  = rev ? (T - 1 - s) : s;
        int tn = rev ? (t > 0 ? t - 1 : 0) : (t < T - 1 ? t + 1 : t);
        const float* gstep = gates + (long)t * (16 * 512);

        // prefetch next step's 32KB gate block (256 thr x 128B lines)
        __builtin_prefetch(gates + (long)tn * (16 * 512) + tid * 32, 0, 1);

        // init accumulators with precomputed x-gates (contiguous block)
        v8f acc[4];
#pragma unroll
        for (int nt = 0; nt < 4; nt++) {
            int n = col0 + nt * 16;
#pragma unroll
            for (int v = 0; v < 8; v++) {
                int m = ((lane < 16) ? 0 : 8) + v;     // batch index
                acc[nt][v] = gstep[m * 512 + n];
            }
        }
        // h @ Whh^T : K=128 in 4 WMMA k-steps, A from LDS (ds_load_b128),
        // B from registers
#pragma unroll
        for (int k4 = 0; k4 < 4; k4++) {
            AF a;
            a.h[0] = *(const v8bf*)(&hbf[mrow][k4 * 32 + kh]);
            a.h[1] = *(const v8bf*)(&hbf[mrow][k4 * 32 + kh + 16]);
#pragma unroll
            for (int nt = 0; nt < 4; nt++) {
                acc[nt] = __builtin_amdgcn_wmma_f32_16x16x32_bf16(
                    false, a.v, false, bw[k4][nt].v, (short)0, acc[nt],
                    false, false);
            }
        }
        // stage gates to LDS
#pragma unroll
        for (int nt = 0; nt < 4; nt++) {
            int n = col0 + nt * 16;
#pragma unroll
            for (int v = 0; v < 8; v++) {
                int m = ((lane < 16) ? 0 : 8) + v;
                gbuf[m][n] = acc[nt][v];
            }
        }
        __syncthreads();

        // elementwise gate math: 16*128 elems / 256 threads = 8 each
#pragma unroll
        for (int i = 0; i < 8; i++) {
            int idx = tid * 8 + i;
            int bb = idx >> 7, j = idx & 127;
            float ig = gbuf[bb][j];
            float fg = gbuf[bb][NH + j];
            float gg = gbuf[bb][2 * NH + j];
            float og = gbuf[bb][3 * NH + j];
            float c = sigf(fg) * cst[bb][j] + sigf(ig) * tanhf_fast(gg);
            float h = sigf(og) * tanhf_fast(c);
            cst[bb][j] = c;
            hbf[bb][j] = (__bf16)h;
            Hout[((long)bb * T + t) * 256 + hoff + j] = h;
        }
        __syncthreads();
    }
}

// ---------------------------------------------------------------------------
// Segment mean: block per (b,l), threads over 256 channels (coalesced)
// ---------------------------------------------------------------------------
__global__ void segmean(const float* __restrict__ H1, const int* __restrict__ dur,
                        __bf16* __restrict__ seq)
{
    int bl = blockIdx.x;                 // 0 .. B*L-1
    int b = bl / NL, l = bl % NL;
    __shared__ int sh_sd[2];
    if (threadIdx.x == 0) {
        const int* db = dur + b * NL;
        int st = 0;
        for (int j = 0; j < l; j++) st += db[j];
        sh_sd[0] = st; sh_sd[1] = db[l];
    }
    __syncthreads();
    int start = sh_sd[0], d = sh_sd[1];
    float inv = 1.f / (float)(d > 0 ? d : 1);
    int c = threadIdx.x;                 // 256 threads = 256 channels
    float s = 0.f;
    for (int k = 0; k < d; k++) {
        int t = start + k;
        if (t < NT) s += H1[((long)b * NT + t) * 256 + c];
    }
    __bf16* o = seq + (long)bl * 288;
    o[c] = (__bf16)((d > 0) ? s * inv : 0.f);
    if (c == 0)  o[256] = (__bf16)(float)d;
    if (c < 31)  o[257 + c] = (__bf16)0.f;
}

// ---------------------------------------------------------------------------
// Host orchestration
// ---------------------------------------------------------------------------
static inline int cdivi(long a, long b) { return (int)((a + b - 1) / b); }

extern "C" void kernel_launch(void* const* d_in, const int* in_sizes, int n_in,
                              void* d_out, int out_size, void* d_ws, size_t ws_size,
                              hipStream_t stream)
{
    (void)in_sizes; (void)n_in; (void)out_size; (void)ws_size;
    const int F[7] = {2, 1024, 512, 512, 256, 256, 128};
    const long ROWS = (long)NB * NT;   // 32768

    const float* pitch  = (const float*)d_in[0];
    const float* energy = (const float*)d_in[1];
    const int*   dur    = (const int*)d_in[3];

    // ---- workspace layout ----
    char* ws = (char*)d_ws;
    const size_t SZ_Y   = (size_t)ROWS * 1024 * 4;                   // 134 MB
    const size_t SZ_ACT = ((size_t)NB * TPAD * 1024 * 2 + 4095) & ~(size_t)4095;
    float*  Y      = (float*)(ws + 0);
    float*  gatesF = (float*)(ws + 0);                               // alias Y
    float*  gatesR = (float*)(ws + (size_t)ROWS * 512 * 4);          // alias Y hi
    const size_t OFF_ACTA = SZ_Y;
    __bf16* actA   = (__bf16*)(ws + OFF_ACTA);
    float*  H1     = (float*)(ws + OFF_ACTA);                        // alias actA
    __bf16* seq    = (__bf16*)(ws + OFF_ACTA + (size_t)ROWS * 256 * 4);
    const size_t OFF_ACTB = OFF_ACTA + SZ_ACT;
    __bf16* actB   = (__bf16*)(ws + OFF_ACTB);
    float*  H2     = (float*)(ws + OFF_ACTB);                        // alias actB
    __bf16* H2B    = (__bf16*)(ws + OFF_ACTB + (size_t)NB * NL * 256 * 4);
    const size_t OFF_W = OFF_ACTB + SZ_ACT;
    __bf16* wreg   = (__bf16*)(ws + OFF_W);
    float*  bnScale = (float*)(ws + OFF_W + (size_t)16 * 1024 * 1024);
    float*  bnShift = bnScale + 1024;

    size_t wo = 0;
    __bf16* convW[6];
    for (int i = 1; i < 6; i++) { convW[i] = wreg + wo; wo += (size_t)F[i + 1] * F[i] * 3; }
    __bf16* wih1f = wreg + wo; wo += 512 * 128;
    __bf16* whh1f = wreg + wo; wo += 512 * 128;
    __bf16* wih1r = wreg + wo; wo += 512 * 128;
    __bf16* whh1r = wreg + wo; wo += 512 * 128;
    __bf16* wih2f = wreg + wo; wo += 512 * 288;
    __bf16* whh2f = wreg + wo; wo += 512 * 128;
    __bf16* wih2r = wreg + wo; wo += 512 * 288;
    __bf16* whh2r = wreg + wo; wo += 512 * 128;
    __bf16* outwb = wreg + wo; wo += 128 * 256;

    // ---- weight conversion (bf16, GEMM-friendly layouts) ----
    for (int i = 1; i < 6; i++) {
        long tot = (long)F[i + 1] * F[i] * 3;
        convw_reorder<<<cdivi(tot, 256), 256, 0, stream>>>(
            (const float*)d_in[5 + 2 * i], convW[i], F[i + 1], F[i]);
    }
    cvt_pad<<<cdivi(512 * 128, 256), 256, 0, stream>>>((const float*)d_in[29], wih1f, 512, 128, 128);
    cvt_pad<<<cdivi(512 * 128, 256), 256, 0, stream>>>((const float*)d_in[30], whh1f, 512, 128, 128);
    cvt_pad<<<cdivi(512 * 128, 256), 256, 0, stream>>>((const float*)d_in[32], wih1r, 512, 128, 128);
    cvt_pad<<<cdivi(512 * 128, 256), 256, 0, stream>>>((const float*)d_in[33], whh1r, 512, 128, 128);
    cvt_pad<<<cdivi(512 * 288, 256), 256, 0, stream>>>((const float*)d_in[35], wih2f, 512, 257, 288);
    cvt_pad<<<cdivi(512 * 128, 256), 256, 0, stream>>>((const float*)d_in[36], whh2f, 512, 128, 128);
    cvt_pad<<<cdivi(512 * 288, 256), 256, 0, stream>>>((const float*)d_in[38], wih2r, 512, 257, 288);
    cvt_pad<<<cdivi(512 * 128, 256), 256, 0, stream>>>((const float*)d_in[39], whh2r, 512, 128, 128);
    cvt_pad<<<cdivi(128 * 256, 256), 256, 0, stream>>>((const float*)d_in[41], outwb, 128, 256, 256);

    // ---- conv stack ----
    conv0_kernel<<<(int)ROWS, 256, 0, stream>>>(
        pitch, energy, (const float*)d_in[5], (const float*)d_in[6], Y);
    bn_stats<<<1024 / 64, 256, 0, stream>>>(Y, (int)ROWS, 1024,
        (const float*)d_in[17], (const float*)d_in[18], bnScale, bnShift);
    zero_pads<<<cdivi(32L * 1024, 256), 256, 0, stream>>>(actA, 1024);
    bn_apply_relu<<<cdivi(ROWS * 1024, 256), 256, 0, stream>>>(
        Y, bnScale, bnShift, actA, ROWS * 1024, 1024);

    __bf16* actIn = actA;
    __bf16* actOut = actB;
    for (int i = 1; i < 6; i++) {
        int Cin = F[i], Cout = F[i + 1];
        int blocks = (int)((ROWS / 16) * (Cout / 64) / 8);
        wmma_gemm<<<blocks, 256, 0, stream>>>(
            actIn, convW[i], (const float*)d_in[6 + 2 * i], Y,
            NT, TPAD, Cin, Cout, 3, 0);
        bn_stats<<<Cout / 64, 256, 0, stream>>>(Y, (int)ROWS, Cout,
            (const float*)d_in[17 + 2 * i], (const float*)d_in[18 + 2 * i], bnScale, bnShift);
        zero_pads<<<cdivi(32L * Cout, 256), 256, 0, stream>>>(actOut, Cout);
        bn_apply_relu<<<cdivi(ROWS * Cout, 256), 256, 0, stream>>>(
            Y, bnScale, bnShift, actOut, ROWS * Cout, Cout);
        __bf16* tmp = actIn; actIn = actOut; actOut = tmp;
    }
    // actIn == actB holds final conv activation, padded [B][T+2][128] bf16

    // ---- LSTM1: x-gates (WMMA GEMM, time-major out), then recurrence ----
    {
        int blocks = (int)((ROWS / 16) * (512 / 64) / 8);   // 2048
        wmma_gemm<<<blocks, 256, 0, stream>>>(actIn + 128, wih1f, (const float*)d_in[31],
                                              gatesF, NT, TPAD, 128, 512, 1, 1);
        wmma_gemm<<<blocks, 256, 0, stream>>>(actIn + 128, wih1r, (const float*)d_in[34],
                                              gatesR, NT, TPAD, 128, 512, 1, 1);
    }
    lstm_rec<<<2, 256, 0, stream>>>(gatesF, gatesR, whh1f, whh1r, H1, NT);

    // ---- segment mean + dur append ----
    segmean<<<NB * NL, 256, 0, stream>>>(H1, dur, seq);

    // ---- LSTM2 ----
    {
        long rows2 = (long)NB * NL;                          // 8192
        int blocks = (int)((rows2 / 16) * (512 / 64) / 8);   // 512
        wmma_gemm<<<blocks, 256, 0, stream>>>(seq, wih2f, (const float*)d_in[37],
                                              gatesF, NL, NL, 288, 512, 1, 1);
        wmma_gemm<<<blocks, 256, 0, stream>>>(seq, wih2r, (const float*)d_in[40],
                                              gatesR, NL, NL, 288, 512, 1, 1);
    }
    lstm_rec<<<2, 256, 0, stream>>>(gatesF, gatesR, whh2f, whh2r, H2, NL);

    // ---- output projection: [8192,256] x [256,128] + b -> d_out f32 ----
    cvt_pad<<<cdivi((long)NB * NL * 256, 256), 256, 0, stream>>>(
        H2, H2B, NB * NL, 256, 256);
    {
        long rows2 = (long)NB * NL;
        int blocks = (int)((rows2 / 16) * (128 / 64) / 8);   // 128
        wmma_gemm<<<blocks, 256, 0, stream>>>(H2B, outwb, (const float*)d_in[42],
                                              (float*)d_out, NL, NL, 256, 128,
                                              1, 0);
    }
}
